// SpatailMamaba_66236985639733
// MI455X (gfx1250) — matrix operand order repaired and checked by reading, not compiled, not run on hardware
//
#include <hip/hip_runtime.h>

typedef __attribute__((ext_vector_type(16))) _Float16 v16h;
typedef __attribute__((ext_vector_type(8)))  float    v8f;
typedef __attribute__((ext_vector_type(4)))  unsigned int v4u;
typedef __attribute__((ext_vector_type(8)))  int      v8i;
typedef __attribute__((ext_vector_type(4)))  int      v4i;

#define HH 128
#define WW 128
#define LL (HH*WW)

__device__ __forceinline__ v8f wmma16x16x32(v16h a, v16h b, v8f c) {
  // D = A(16x32 f16) * B(32x16 f16) + C(16x16 f32)
  return __builtin_amdgcn_wmma_f32_16x16x32_f16(false, a, false, b, (short)0, c, false, false);
}

// Scan-direction index transforms (H == W == 128):
//  t=0: identity    t=1: l=w*H+h -> pixel h*W+w   t=2: flip   t=3: flip of t=1
__device__ __forceinline__ int mtrans(int m, int t) {
  switch (t) {
    default:
    case 0: return m;
    case 1: return (m & 127) * WW + (m >> 7);
    case 2: return LL - 1 - m;
    case 3: { int l = LL - 1 - m; return (l & 127) * WW + (l >> 7); }
  }
}

// ---------------------------------------------------------------------------
// Tensor Data Mover: DMA a [rows x 16] fp32 tile (row stride = lda elements)
// from global memory into LDS (row-major, contiguous). D# per CDNA5 ISA §8.
// Issued by one wave; completion via TENSORcnt.
// ---------------------------------------------------------------------------
__device__ __forceinline__ void tdm_load_tile_2d(const float* gsrc, void* ldsdst,
                                                 int rows, int lda) {
  unsigned long long ga = (unsigned long long)(uintptr_t)gsrc;
  v4u g0;
  g0[0] = 1u;                                            // count=1, user mode
  g0[1] = (unsigned)(uintptr_t)ldsdst;                   // lds_addr (bytes)
  g0[2] = (unsigned)ga;                                  // global_addr[31:0]
  g0[3] = ((unsigned)(ga >> 32) & 0x01FFFFFFu) | (2u << 30); // addr[56:32] | type=2
  unsigned td0 = 16u, td1 = (unsigned)rows, st = (unsigned)lda;
  v8i g1;
  g1[0] = (int)0x20000u;                                 // data_size = 2 (4 bytes)
  g1[1] = (int)((td0 & 0xFFFFu) << 16);                  // tensor_dim0[15:0]
  g1[2] = (int)((td0 >> 16) | ((td1 & 0xFFFFu) << 16));  // tensor_dim0 hi | tensor_dim1 lo
  g1[3] = (int)((td1 >> 16) | (16u << 16));              // tensor_dim1 hi | tile_dim0=16
  g1[4] = (int)(td1 & 0xFFFFu);                          // tile_dim1 = rows (tile_dim2=0)
  g1[5] = (int)st;                                       // tensor_dim0_stride[31:0]
  g1[6] = 0;                                             // stride hi | dim1_stride lo
  g1[7] = 0;
  v4i z4 = {0, 0, 0, 0};
#if defined(__clang_major__) && __clang_major__ >= 23
  v8i z8 = {0, 0, 0, 0, 0, 0, 0, 0};
  __builtin_amdgcn_tensor_load_to_lds(g0, g1, z4, z4, z8, 0);
#else
  __builtin_amdgcn_tensor_load_to_lds(g0, g1, z4, z4, 0);
#endif
}

// ---------------------------------------------------------------------------
// Pack fp32 weights into f16 B-fragments, exact per-lane WMMA layout:
// dst[tap][ntile][kstep][lane(32)][j(16)], element = B[k][n] with
// k = kstep*32 + (lane>>4)*8 + (j&7) + (j>>3)*16, n = ntile*16 + (lane&15),
// B[k][n] = src[n*sn + k*sk + tap*st], zero-padded outside K,N.
// ---------------------------------------------------------------------------
__global__ void pack_b_kernel(const float* __restrict__ src, _Float16* __restrict__ dst,
                              int taps, int ntiles, int ksteps,
                              int K, int N, int sn, int sk, int st) {
  int total = taps * ntiles * ksteps * 32 * 16;
  for (int i = blockIdx.x * blockDim.x + threadIdx.x; i < total; i += gridDim.x * blockDim.x) {
    int j = i & 15;
    int lane = (i >> 4) & 31;
    int t = i >> 9;
    int kstep = t % ksteps; t /= ksteps;
    int ntile = t % ntiles; t /= ntiles;
    int tap = t;
    int n = ntile * 16 + (lane & 15);
    int k = kstep * 32 + ((lane >> 4) * 8) + (j & 7) + ((j >> 3) << 4);
    float v = 0.f;
    if (k < K && n < N) v = src[(size_t)n * sn + (size_t)k * sk + (size_t)tap * st];
    dst[i] = (_Float16)v;
  }
}

// ---------------------------------------------------------------------------
// Implicit-GEMM dense conv (RxR, 64 cin, 64 cout) with WMMA.
// Block = 128 threads (4 waves). M tile = 16 consecutive x at one row y.
// Wave w computes couts [16w,16w+16). Epilogue: relu(acc+bias) -> CHW out.
// ---------------------------------------------------------------------------
template<int R>
__global__ void __launch_bounds__(128) conv_wmma_kernel(
    const float* __restrict__ x, const _Float16* __restrict__ wfrag,
    const float* __restrict__ bias, float* __restrict__ out, int choff) {
  constexpr int PADW = 16 + R - 1;
  constexpr int HALO = R / 2;
  __shared__ float lds[64][R][PADW];
  const int y0 = blockIdx.x >> 3;          // W/16 = 8 tiles per row
  const int x0 = (blockIdx.x & 7) << 4;
  for (int i = threadIdx.x; i < 64 * R * PADW; i += 128) {
    int px = i % PADW;
    int ry = (i / PADW) % R;
    int c  = i / (PADW * R);
    int gy = y0 + ry - HALO, gx = x0 + px - HALO;
    float v = 0.f;
    if (gy >= 0 && gy < HH && gx >= 0 && gx < WW) v = x[((size_t)c * HH + gy) * WW + gx];
    lds[c][ry][px] = v;
  }
  __syncthreads();
  const int lane = threadIdx.x & 31;
  const int wv   = threadIdx.x >> 5;
  const int mrow = lane & 15;
  const int ko   = (lane >> 4) << 3;
  v8f acc = {};
#pragma unroll(R == 3 ? 9 : 1)
  for (int tap = 0; tap < R * R; ++tap) {
    const int dy = tap / R, dx = tap % R;
#pragma unroll
    for (int ks = 0; ks < 2; ++ks) {
      v16h a;
#pragma unroll
      for (int j = 0; j < 8; ++j) {
        a[j]     = (_Float16)lds[ks * 32 + ko + j][dy][mrow + dx];
        a[8 + j] = (_Float16)lds[ks * 32 + 16 + ko + j][dy][mrow + dx];
      }
      v16h b = *(const v16h*)(wfrag + ((((size_t)(tap * 4 + wv) * 2 + ks) * 32 + lane) << 4));
      acc = wmma16x16x32(a, b, acc);
    }
  }
  const int n  = (wv << 4) + (lane & 15);
  const int mo = (lane >> 4) << 3;
  const float bn = bias[n];
  float r[8];
#pragma unroll
  for (int i = 0; i < 8; ++i) { float v = acc[i] + bn; r[i] = v > 0.f ? v : 0.f; }
  float* op = out + (size_t)(choff + n) * LL + (size_t)y0 * WW + x0 + mo;
  *(float4*)(op)     = make_float4(r[0], r[1], r[2], r[3]);
  *(float4*)(op + 4) = make_float4(r[4], r[5], r[6], r[7]);
}

// ---------------------------------------------------------------------------
// Generic WMMA GEMM, A is K-major (A[k][m], row stride lda) with optional
// column index transform; B pre-packed fragments; out N-major (out[n][M]).
// KSTEPS = K/32 (compile-time: statically indexed fragments, no v_movrels).
// ACT: 0=none 1=relu. addsrc added after activation.
// TDM=true: stage the A tile with the Tensor Data Mover (trans must be 0).
// ---------------------------------------------------------------------------
template<int KSTEPS, int ACT, bool TDM>
__global__ void __launch_bounds__(128) gemm_kmajor_kernel(
    const float* __restrict__ A, int lda,
    const _Float16* __restrict__ wfrag,
    const float* __restrict__ bias,
    const float* __restrict__ addsrc,
    float* __restrict__ out,
    int M, int ntiles, int nvalid, int trans) {
  constexpr int K = KSTEPS * 32;
  __shared__ float lds[K][16];
  const int m0 = blockIdx.x << 4;
  if constexpr (TDM) {
    if (threadIdx.x < 32) {
      tdm_load_tile_2d(A + m0, &lds[0][0], K, lda);
      __builtin_amdgcn_s_wait_tensorcnt(0);
    }
    __syncthreads();
  } else {
    for (int i = threadIdx.x; i < (K << 4); i += 128) {
      int mi = i & 15, k = i >> 4;
      lds[k][mi] = A[(size_t)k * lda + mtrans(m0 + mi, trans)];
    }
    __syncthreads();
  }
  const int lane = threadIdx.x & 31;
  const int wv   = threadIdx.x >> 5;
  const int mrow = lane & 15;
  const int ko   = (lane >> 4) << 3;
  v16h afrag[KSTEPS];
#pragma unroll
  for (int ks = 0; ks < KSTEPS; ++ks) {
#pragma unroll
    for (int j = 0; j < 8; ++j) {
      afrag[ks][j]     = (_Float16)lds[ks * 32 + ko + j][mrow];
      afrag[ks][8 + j] = (_Float16)lds[ks * 32 + 16 + ko + j][mrow];
    }
  }
  const int n16 = lane & 15;
  const int mo  = (lane >> 4) << 3;
  for (int nt = wv; nt < ntiles; nt += 4) {
    v8f acc = {};
#pragma unroll
    for (int ks = 0; ks < KSTEPS; ++ks) {
      v16h b = *(const v16h*)(wfrag + (((size_t)(nt * KSTEPS + ks) * 32 + lane) << 4));
      acc = wmma16x16x32(afrag[ks], b, acc);
    }
    const int n = (nt << 4) + n16;
    if (n < nvalid) {
      float bn = bias ? bias[n] : 0.f;
      float r[8];
#pragma unroll
      for (int i = 0; i < 8; ++i) {
        float v = acc[i] + bn;
        if (ACT == 1) v = v > 0.f ? v : 0.f;
        r[i] = v;
      }
      size_t o = (size_t)n * M + m0 + mo;
      if (addsrc) {
#pragma unroll
        for (int i = 0; i < 8; ++i) r[i] += addsrc[o + i];
      }
      *(float4*)(out + o)     = make_float4(r[0], r[1], r[2], r[3]);
      *(float4*)(out + o + 4) = make_float4(r[4], r[5], r[6], r[7]);
    }
  }
}

// ---------------------------------------------------------------------------
// Grouped conv 3x3, groups=64, 2 in / 2 out per group (dw1a).
// ---------------------------------------------------------------------------
__global__ void dwgroup2_kernel(const float* __restrict__ in, const float* __restrict__ w,
                                const float* __restrict__ b, float* __restrict__ out) {
  int idx = blockIdx.x * blockDim.x + threadIdx.x;
  if (idx >= 128 * LL) return;
  int p = idx & (LL - 1);
  int co = idx >> 14;
  int g2 = (co >> 1) << 1;
  int py = p >> 7, px = p & 127;
  float acc = b[co];
  for (int ci = 0; ci < 2; ++ci) {
    const float* ip = in + (size_t)(g2 + ci) * LL;
    const float* wp = w + (co * 2 + ci) * 9;
#pragma unroll
    for (int t = 0; t < 9; ++t) {
      int yy = py + t / 3 - 1, xx = px + t % 3 - 1;
      if (yy >= 0 && yy < HH && xx >= 0 && xx < WW) acc += wp[t] * ip[yy * WW + xx];
    }
  }
  out[idx] = acc;
}

// Depthwise conv 3x3 (groups == C), act: 0=none, 2=silu.
__global__ void dwconv3_kernel(const float* __restrict__ in, const float* __restrict__ w,
                               const float* __restrict__ b, float* __restrict__ out,
                               int C, int act) {
  int idx = blockIdx.x * blockDim.x + threadIdx.x;
  if (idx >= C * LL) return;
  int p = idx & (LL - 1);
  int c = idx >> 14;
  int py = p >> 7, px = p & 127;
  const float* ip = in + (size_t)c * LL;
  const float* wp = w + c * 9;
  float acc = b[c];
#pragma unroll
  for (int t = 0; t < 9; ++t) {
    int yy = py + t / 3 - 1, xx = px + t % 3 - 1;
    if (yy >= 0 && yy < HH && xx >= 0 && xx < WW) acc += wp[t] * ip[yy * WW + xx];
  }
  if (act == 2) acc = acc / (1.f + __expf(-acc));
  out[idx] = acc;
}

// ---------------------------------------------------------------------------
// Global instance-norm reductions.
// ---------------------------------------------------------------------------
__global__ void redsum_kernel(const float* __restrict__ x, int n, float* __restrict__ partial) {
  __shared__ float s1[256], s2[256];
  float a = 0.f, b = 0.f;
  for (int i = blockIdx.x * 256 + threadIdx.x; i < n; i += gridDim.x * 256) {
    float v = x[i]; a += v; b += v * v;
  }
  s1[threadIdx.x] = a; s2[threadIdx.x] = b;
  __syncthreads();
  for (int s = 128; s > 0; s >>= 1) {
    if (threadIdx.x < (unsigned)s) {
      s1[threadIdx.x] += s1[threadIdx.x + s];
      s2[threadIdx.x] += s2[threadIdx.x + s];
    }
    __syncthreads();
  }
  if (threadIdx.x == 0) { partial[blockIdx.x * 2] = s1[0]; partial[blockIdx.x * 2 + 1] = s2[0]; }
}

__global__ void finalize_kernel(const float* __restrict__ partial, int nb, float n,
                                float* __restrict__ stats) {
  __shared__ float s1[256], s2[256];
  float a = 0.f, b = 0.f;
  for (int i = threadIdx.x; i < nb; i += 256) { a += partial[i * 2]; b += partial[i * 2 + 1]; }
  s1[threadIdx.x] = a; s2[threadIdx.x] = b;
  __syncthreads();
  for (int s = 128; s > 0; s >>= 1) {
    if (threadIdx.x < (unsigned)s) {
      s1[threadIdx.x] += s1[threadIdx.x + s];
      s2[threadIdx.x] += s2[threadIdx.x + s];
    }
    __syncthreads();
  }
  if (threadIdx.x == 0) {
    float mu = s1[0] / n;
    float var = s2[0] / n - mu * mu;
    stats[0] = mu;
    stats[1] = rsqrtf(var + 1e-5f);
  }
}

__global__ void norm_apply_kernel(const float* __restrict__ x, const float* __restrict__ stats,
                                  float* __restrict__ out, int n) {
  int i = blockIdx.x * blockDim.x + threadIdx.x;
  if (i < n) out[i] = (x[i] - stats[0]) * stats[1];
}

// Per-pixel layernorm over channels (CHW layout, coalesced over pixels).
__global__ void ln_chw_kernel(const float* __restrict__ x, const float* __restrict__ g,
                              const float* __restrict__ b, float* __restrict__ out, int C) {
  int p = blockIdx.x * blockDim.x + threadIdx.x;
  if (p >= LL) return;
  float mu = 0.f;
  for (int c = 0; c < C; ++c) mu += x[(size_t)c * LL + p];
  mu /= (float)C;
  float var = 0.f;
  for (int c = 0; c < C; ++c) { float d = x[(size_t)c * LL + p] - mu; var += d * d; }
  var /= (float)C;
  float rs = rsqrtf(var + 1e-5f);
  for (int c = 0; c < C; ++c)
    out[(size_t)c * LL + p] = (x[(size_t)c * LL + p] - mu) * rs * g[c] + b[c];
}

// out[c][p] = LN_128(y)[c][p] * silu(z[c][p])  (A-matrix for out_proj GEMM)
__global__ void ln_silu_kernel(const float* __restrict__ y, const float* __restrict__ z,
                               const float* __restrict__ g, const float* __restrict__ b,
                               float* __restrict__ out) {
  int p = blockIdx.x * blockDim.x + threadIdx.x;
  if (p >= LL) return;
  float mu = 0.f;
  for (int c = 0; c < 128; ++c) mu += y[(size_t)c * LL + p];
  mu *= (1.f / 128.f);
  float var = 0.f;
  for (int c = 0; c < 128; ++c) { float d = y[(size_t)c * LL + p] - mu; var += d * d; }
  var *= (1.f / 128.f);
  float rs = rsqrtf(var + 1e-5f);
  for (int c = 0; c < 128; ++c) {
    float zz = z[(size_t)c * LL + p];
    float s = zz / (1.f + __expf(-zz));
    out[(size_t)c * LL + p] = ((y[(size_t)c * LL + p] - mu) * rs * g[c] + b[c]) * s;
  }
}

// delta[k][d][l] = softplus(sum_r dts[k][r][l]*dt_w[k][d][r] + dt_b[k][d])
__global__ void delta_kernel(const float* __restrict__ xdbl, const float* __restrict__ dtw,
                             const float* __restrict__ dtb, float* __restrict__ delta) {
  int idx = blockIdx.x * blockDim.x + threadIdx.x;
  if (idx >= 4 * 128 * LL) return;
  int l = idx & (LL - 1);
  int d = (idx >> 14) & 127;
  int k = idx >> 21;
  const float* dts = xdbl + (size_t)k * 36 * LL;   // rows 0..3 = dts
  const float* wp = dtw + (k * 128 + d) * 4;
  float acc = dtb[k * 128 + d];
#pragma unroll
  for (int r = 0; r < 4; ++r) acc += dts[(size_t)r * LL + l] * wp[r];
  delta[idx] = (acc > 20.f) ? acc : log1pf(__expf(acc));
}

// ---------------------------------------------------------------------------
// Chunked parallel scan: L=16384 split into 128 chunks of 128.
// h_l = exp(delta*A_n) * h_{l-1} + delta*u*B_n ; y_l = sum_n C_n h_n
// ---------------------------------------------------------------------------
__global__ void scan_phase1_kernel(const float* __restrict__ delta, const float* __restrict__ xa,
                                   const float* __restrict__ xdbl, const float* __restrict__ alog,
                                   float* __restrict__ sumdelta, float* __restrict__ hpart) {
  int idx = blockIdx.x * blockDim.x + threadIdx.x;
  if (idx >= 4 * 128 * 128) return;
  int c = idx & 127;
  int d = (idx >> 7) & 127;
  int k = idx >> 14;
  const float* del = delta + (size_t)(k * 128 + d) * LL + c * 128;
  const float* Bp  = xdbl + (size_t)k * 36 * LL + (size_t)4 * LL + c * 128; // rows 4..19 = Bs
  const float* xd  = xa + (size_t)d * LL;
  float An[16];
#pragma unroll
  for (int n = 0; n < 16; ++n) An[n] = -__expf(alog[(k * 128 + d) * 16 + n]);
  float h[16];
#pragma unroll
  for (int n = 0; n < 16; ++n) h[n] = 0.f;
  float sd = 0.f;
  for (int t = 0; t < 128; ++t) {
    float dl = del[t];
    sd += dl;
    float u = xd[mtrans(c * 128 + t, k)];
    float du = dl * u;
#pragma unroll
    for (int n = 0; n < 16; ++n)
      h[n] = h[n] * __expf(dl * An[n]) + du * Bp[(size_t)n * LL + t];
  }
  sumdelta[idx] = sd;
#pragma unroll
  for (int n = 0; n < 16; ++n) hpart[(size_t)idx * 16 + n] = h[n];
}

__global__ void scan_phase2_kernel(const float* __restrict__ sumdelta,
                                   const float* __restrict__ hpart,
                                   const float* __restrict__ alog, float* __restrict__ h0) {
  int idx = blockIdx.x * blockDim.x + threadIdx.x;
  if (idx >= 4 * 128 * 16) return;
  int n = idx & 15;
  int d = (idx >> 4) & 127;
  int k = idx >> 11;
  float An = -__expf(alog[(k * 128 + d) * 16 + n]);
  int base = (k * 128 + d) * 128;
  float h = 0.f;
  for (int c = 0; c < 128; ++c) {
    h0[(size_t)(base + c) * 16 + n] = h;
    h = h * __expf(An * sumdelta[base + c]) + hpart[(size_t)(base + c) * 16 + n];
  }
}

__global__ void scan_phase3_kernel(const float* __restrict__ delta, const float* __restrict__ xa,
                                   const float* __restrict__ xdbl, const float* __restrict__ alog,
                                   const float* __restrict__ h0, float* __restrict__ ys) {
  int idx = blockIdx.x * blockDim.x + threadIdx.x;
  if (idx >= 4 * 128 * 128) return;
  int c = idx & 127;
  int d = (idx >> 7) & 127;
  int k = idx >> 14;
  const float* del = delta + (size_t)(k * 128 + d) * LL + c * 128;
  const float* Bp  = xdbl + (size_t)k * 36 * LL + (size_t)4 * LL + c * 128;
  const float* Cp  = xdbl + (size_t)k * 36 * LL + (size_t)20 * LL + c * 128; // rows 20..35 = Cs
  const float* xd  = xa + (size_t)d * LL;
  float An[16];
#pragma unroll
  for (int n = 0; n < 16; ++n) An[n] = -__expf(alog[(k * 128 + d) * 16 + n]);
  float h[16];
#pragma unroll
  for (int n = 0; n < 16; ++n) h[n] = h0[(size_t)idx * 16 + n];
  float* yp = ys + (size_t)(k * 128 + d) * LL + c * 128;
  for (int t = 0; t < 128; ++t) {
    float dl = del[t];
    float u = xd[mtrans(c * 128 + t, k)];
    float du = dl * u;
    float acc = 0.f;
#pragma unroll
    for (int n = 0; n < 16; ++n) {
      h[n] = h[n] * __expf(dl * An[n]) + du * Bp[(size_t)n * LL + t];
      acc += h[n] * Cp[(size_t)n * LL + t];
    }
    yp[t] = acc;
  }
}

// y[d][p] = ys0[p] + ys1[swap(p)] + ys2[L-1-p] + ys3[swap(L-1-p)] + x0[d][p]*sum_k D[k][d]
__global__ void combine_kernel(const float* __restrict__ ys, const float* __restrict__ xa,
                               const float* __restrict__ Ds, float* __restrict__ ycomb) {
  int idx = blockIdx.x * blockDim.x + threadIdx.x;
  if (idx >= 128 * LL) return;
  int p = idx & (LL - 1);
  int d = idx >> 14;
  float dsum = Ds[d] + Ds[128 + d] + Ds[256 + d] + Ds[384 + d];
  float acc = xa[idx] * dsum;
  acc += ys[(size_t)(0 * 128 + d) * LL + p];
  acc += ys[(size_t)(1 * 128 + d) * LL + mtrans(p, 1)];
  acc += ys[(size_t)(2 * 128 + d) * LL + (LL - 1 - p)];
  acc += ys[(size_t)(3 * 128 + d) * LL + mtrans(p, 3)];
  ycomb[idx] = acc;
}

// ---------------------------------------------------------------------------
extern "C" void kernel_launch(void* const* d_in, const int* in_sizes, int n_in,
                              void* d_out, int out_size, void* d_ws, size_t ws_size,
                              hipStream_t stream) {
  const float* x         = (const float*)d_in[0];
  const float* conv1_w   = (const float*)d_in[1];
  const float* conv1_b   = (const float*)d_in[2];
  const float* conv2_w   = (const float*)d_in[3];
  const float* conv2_b   = (const float*)d_in[4];
  const float* dw1a_w    = (const float*)d_in[5];
  const float* dw1a_b    = (const float*)d_in[6];
  const float* dw1b_w    = (const float*)d_in[7];
  const float* dw1b_b    = (const float*)d_in[8];
  const float* dw2a_w    = (const float*)d_in[9];
  const float* dw2a_b    = (const float*)d_in[10];
  const float* dw2b_w    = (const float*)d_in[11];
  const float* dw2b_b    = (const float*)d_in[12];
  const float* ln_g      = (const float*)d_in[13];
  const float* ln_b      = (const float*)d_in[14];
  const float* in_proj_w = (const float*)d_in[15];
  const float* conv_dw_w = (const float*)d_in[16];
  const float* conv_dw_b = (const float*)d_in[17];
  const float* x_proj_w  = (const float*)d_in[18];
  const float* dt_w      = (const float*)d_in[19];
  const float* dt_b      = (const float*)d_in[20];
  const float* A_logs    = (const float*)d_in[21];
  const float* Ds        = (const float*)d_in[22];
  const float* on_g      = (const float*)d_in[23];
  const float* on_b      = (const float*)d_in[24];
  const float* out_w     = (const float*)d_in[25];

  char* ws = (char*)d_ws;
  // packed f16 weight fragments (~0.4 MB)
  _Float16* P_C1   = (_Float16*)(ws + 0);        //  9*4*2*512 h
  _Float16* P_C2   = (_Float16*)(ws + 73728);    // 25*4*2*512 h
  _Float16* P_DW1B = (_Float16*)(ws + 278528);   //  4*4*512 h
  _Float16* P_INPJ = (_Float16*)(ws + 294912);   // 16*2*512 h
  _Float16* P_XP   = (_Float16*)(ws + 327680);   //  4*3*4*512 h
  _Float16* P_OUTW = (_Float16*)(ws + 376832);   //  4*4*512 h
  _Float16* P_DW2B = (_Float16*)(ws + 393216);   //  4*2*512 h
  float* partial = (float*)(ws + 401408);        // 512 floats
  float* stats   = (float*)(ws + 403456);        // mean, rstd
  // activations (floats); ~140 MB total with slot reuse
  float* F       = (float*)(ws + 458752);
  float* f_ybig  = F;                 // 128*L  concat conv out
  float* f_dw1a  = F + 2097152;       // 128*L
  float* f_ydw   = F + 4194304;       //  64*L
  float* f_ynorm = F + 5242880;       //  64*L
  float* f_lnout = F + 6291456;       //  64*L
  float* f_xz    = F + 7340032;       // 256*L (xa rows 0..127, z rows 128..255)
  float* f_z     = f_xz + 2097152;
  float* f_xas   = F + 11534336;      // 128*L  silu(dwconv(xa))
  float* f_xdbl  = F + 13631488;      // 4*36*L
  float* f_delta = F + 15990784;      // 4*128*L
  float* f_sumd  = F + 24379392;      // 4*128*128
  float* f_hpart = F + 24444928;      // 4*128*128*16
  float* f_h0    = F + 25493504;      // 4*128*128*16
  float* f_ys    = F + 26542080;      // 4*128*L
  float* f_ycomb = f_ybig;            // reuse (ybig dead after dw1a)
  float* f_A2    = f_dw1a;            // reuse (dw1a_out dead after dw1b)
  float* f_y2    = f_ydw;             // reuse (ydw dead after instnorm)
  float* f_t1    = f_lnout;           // reuse (lnout dead after in_proj)

  // ---- pack all WMMA B-operands to f16 fragment order ----
  pack_b_kernel<<<64, 256, 0, stream>>>(conv1_w,   P_C1,   9, 4, 2, 64, 64, 576, 9, 1);
  pack_b_kernel<<<64, 256, 0, stream>>>(conv2_w,   P_C2,  25, 4, 2, 64, 64, 1600, 25, 1);
  pack_b_kernel<<<16, 256, 0, stream>>>(dw1b_w,    P_DW1B, 1, 4, 4, 128, 64, 128, 1, 0);
  pack_b_kernel<<<32, 256, 0, stream>>>(in_proj_w, P_INPJ, 1, 16, 2, 64, 256, 64, 1, 0);
  pack_b_kernel<<<48, 256, 0, stream>>>(x_proj_w,  P_XP,   4, 3, 4, 128, 36, 128, 1, 36 * 128);
  pack_b_kernel<<<16, 256, 0, stream>>>(out_w,     P_OUTW, 1, 4, 4, 128, 64, 128, 1, 0);
  pack_b_kernel<<< 8, 256, 0, stream>>>(dw2b_w,    P_DW2B, 1, 4, 2, 64, 64, 64, 1, 0);

  // ---- front convs: y = relu(concat(conv3x3, conv5x5)) ----
  conv_wmma_kernel<3><<<1024, 128, 0, stream>>>(x, P_C1, conv1_b, f_ybig, 0);
  conv_wmma_kernel<5><<<1024, 128, 0, stream>>>(x, P_C2, conv2_b, f_ybig, 64);

  // ---- dw1a (grouped 3x3) then dw1b (1x1 GEMM, relu), A tile via TDM ----
  dwgroup2_kernel<<<(128 * LL) / 256, 256, 0, stream>>>(f_ybig, dw1a_w, dw1a_b, f_dw1a);
  gemm_kmajor_kernel<4, 1, true><<<1024, 128, 0, stream>>>(f_dw1a, LL, P_DW1B, dw1b_b, nullptr,
                                                           f_ydw, LL, 4, 64, 0);

  // ---- global instance norm ----
  redsum_kernel<<<256, 256, 0, stream>>>(f_ydw, 64 * LL, partial);
  finalize_kernel<<<1, 256, 0, stream>>>(partial, 256, (float)(64 * LL), stats);
  norm_apply_kernel<<<(64 * LL) / 256, 256, 0, stream>>>(f_ydw, stats, f_ynorm, 64 * LL);

  // ---- SS2D ----
  ln_chw_kernel<<<LL / 256, 256, 0, stream>>>(f_ynorm, ln_g, ln_b, f_lnout, 64);
  gemm_kmajor_kernel<2, 0, true><<<1024, 128, 0, stream>>>(f_lnout, LL, P_INPJ, nullptr, nullptr,
                                                           f_xz, LL, 16, 256, 0);
  dwconv3_kernel<<<(128 * LL) / 256, 256, 0, stream>>>(f_xz, conv_dw_w, conv_dw_b, f_xas, 128, 2);
  // direction 0 is a plain rectangular tile -> TDM; 1..3 use index transforms
  gemm_kmajor_kernel<4, 0, true><<<1024, 128, 0, stream>>>(f_xas, LL, P_XP, nullptr, nullptr,
                                                           f_xdbl, LL, 3, 36, 0);
  for (int k = 1; k < 4; ++k)
    gemm_kmajor_kernel<4, 0, false><<<1024, 128, 0, stream>>>(
        f_xas, LL, P_XP + (size_t)k * 3 * 4 * 512, nullptr, nullptr,
        f_xdbl + (size_t)k * 36 * LL, LL, 3, 36, k);
  delta_kernel<<<(4 * 128 * LL) / 256, 256, 0, stream>>>(f_xdbl, dt_w, dt_b, f_delta);

  scan_phase1_kernel<<<256, 256, 0, stream>>>(f_delta, f_xas, f_xdbl, A_logs, f_sumd, f_hpart);
  scan_phase2_kernel<<<32, 256, 0, stream>>>(f_sumd, f_hpart, A_logs, f_h0);
  scan_phase3_kernel<<<256, 256, 0, stream>>>(f_delta, f_xas, f_xdbl, A_logs, f_h0, f_ys);
  combine_kernel<<<(128 * LL) / 256, 256, 0, stream>>>(f_ys, f_xas, Ds, f_ycomb);

  ln_silu_kernel<<<LL / 256, 256, 0, stream>>>(f_ycomb, f_z, on_g, on_b, f_A2);
  // out_proj fused with "+ ynorm" residual -> y2
  gemm_kmajor_kernel<4, 0, true><<<1024, 128, 0, stream>>>(f_A2, LL, P_OUTW, nullptr, f_ynorm,
                                                           f_y2, LL, 4, 64, 0);

  // ---- tail: dw2a depthwise, dw2b 1x1 GEMM (relu) fused with + x residual ----
  dwconv3_kernel<<<(64 * LL) / 256, 256, 0, stream>>>(f_y2, dw2a_w, dw2a_b, f_t1, 64, 0);
  gemm_kmajor_kernel<2, 1, true><<<1024, 128, 0, stream>>>(f_t1, LL, P_DW2B, dw2b_b, x,
                                                           (float*)d_out, LL, 4, 64, 0);
}
